// LSTMencoder_89747636617779
// MI455X (gfx1250) — matrix operand order repaired and checked by reading, not compiled, not run on hardware
//
#include <hip/hip_runtime.h>
#include <hip/hip_bf16.h>

// ---------------------------------------------------------------------------
// LSTM encoder for MI455X (gfx1250, wave32, WMMA bf16 16x16x32, f32 accum)
//
// B=32, S=512, D=HID=512, LAYERS=2, gate width 4*HID=2048, Kcat=1024
//
// Recurrence runs as a diagonal pipeline: launch t computes layer0 step t
// (WGs 0-7) concurrently with layer1 step t-1 (WGs 8-15). 513 launches total.
// ---------------------------------------------------------------------------

typedef __attribute__((ext_vector_type(16))) __bf16 v16bf;
typedef __attribute__((ext_vector_type(8)))  __bf16 v8bf;
typedef __attribute__((ext_vector_type(8)))  float  v8f;

#define Bdim   32
#define Sdim   512
#define Ddim   512
#define Hdim   512
#define G4     2048   // 4*HID
#define KCAT   1024   // D + HID
#define HS     ((size_t)Bdim * Hdim)     // one state slot / slice
#define WL     ((size_t)KCAT * G4)       // packed weight elems per layer

// ---------------------------------------------------------------------------
// 1) Embedding gather: emb[source] -> fp32 (d_out second half) + bf16 (ws)
// ---------------------------------------------------------------------------
__global__ void lstm_embed_gather(const int* __restrict__ src,
                                  const float* __restrict__ emb,
                                  float* __restrict__ out_emb,
                                  __bf16* __restrict__ Xemb) {
  const int row = blockIdx.x;                 // b*S + s, 0..16383
  const int tok = src[row];
  const float4* e4 = reinterpret_cast<const float4*>(emb + (size_t)tok * Ddim);
  float4*       o4 = reinterpret_cast<float4*>(out_emb + (size_t)row * Ddim);
  __bf16*       xb = Xemb + (size_t)row * Ddim;
  for (int i = threadIdx.x; i < Ddim / 4; i += blockDim.x) {
    float4 v = e4[i];
    o4[i] = v;
    const int k = i * 4;
    xb[k + 0] = (__bf16)v.x;
    xb[k + 1] = (__bf16)v.y;
    xb[k + 2] = (__bf16)v.z;
    xb[k + 3] = (__bf16)v.w;
  }
}

// ---------------------------------------------------------------------------
// 2) Pack [Wk;Wr] -> bf16 fragment-major B-matrix layout.
// Wpack[l][u][g][ki][lane][e], u=unit-tile(32), g=gate(4), ki=kstep(32)
// B frag: lane holds col N=u*16+(lane&15), K = ki*32 + (lane>>4)*16 + e
// Each lane's 16 bf16 are contiguous => coalesced 32B loads in the GEMM.
// ---------------------------------------------------------------------------
__global__ void lstm_pack_weights(const float* __restrict__ Wk,
                                  const float* __restrict__ Wr,
                                  __bf16* __restrict__ Wpack) {
  const int idx  = blockIdx.x * blockDim.x + threadIdx.x;  // 0..4194303
  const int e    = idx & 15;
  const int lane = (idx >> 4) & 31;
  const int ki   = (idx >> 9) & 31;
  const int g    = (idx >> 14) & 3;
  const int u    = (idx >> 16) & 31;
  const int l    = (idx >> 21) & 1;
  const int K    = ki * 32 + (lane >> 4) * 16 + e;         // 0..1023
  const int n    = g * Hdim + u * 16 + (lane & 15);        // 0..2047
  float v;
  if (K < Ddim) v = Wk[((size_t)l * Ddim + K) * G4 + n];
  else          v = Wr[((size_t)l * Hdim + (K - Ddim)) * G4 + n];
  Wpack[idx] = (__bf16)v;
}

// ---------------------------------------------------------------------------
// 3) Zero-init recurrent state (h ping-pong bf16, c fp32)
// ---------------------------------------------------------------------------
__global__ void lstm_init_state(__bf16* __restrict__ hbuf, float* __restrict__ cbuf) {
  const int idx = blockIdx.x * blockDim.x + threadIdx.x;
  if (idx < 2 * 2 * Bdim * Hdim) hbuf[idx] = (__bf16)0.0f;  // 2 layers x 2 slots
  if (idx < 2 * Bdim * Hdim)     cbuf[idx] = 0.0f;
}

// ---------------------------------------------------------------------------
// 4) Fused pipelined step. Launch t:
//    WGs 0-7 : layer 0, step t    (t < S)
//    WGs 8-15: layer 1, step t-1  (t >= 1)
// Each wave u owns hidden units [u*16, u*16+16) across ALL 4 gates so the
// LSTM nonlinearity fuses in registers. Per wave: 2 M-tiles x 4 gate
// accumulators, 32 K-steps, 8 v_wmma_f32_16x16x32_bf16 per K-step
// (A fragments reused across the 4 gates).
// ---------------------------------------------------------------------------
__global__ __launch_bounds__(128) void lstm_pipe_step(
    const __bf16* __restrict__ Xemb,     // [B,S,D] bf16
    __bf16* __restrict__ hbuf,           // [2 layers][2 slots][B,H] bf16
    float* __restrict__ cbuf,            // [2 layers][B,H] fp32 (in place)
    const float* __restrict__ bias,      // [2][2048]
    const __bf16* __restrict__ Wpack,    // [2] packed layer weights
    __bf16* __restrict__ X1buf,          // [2 slots][B,H] inter-layer slices
    float* __restrict__ out_h,           // d_out [B,S,H]
    int t) {
  const int role = blockIdx.x >> 3;      // 0 = layer0, 1 = layer1
  const int wg   = blockIdx.x & 7;

  const __bf16* x_base;
  long x_stride;
  __bf16* x_next;
  float*  out;
  int l, step;
  if (role == 0) {
    if (t >= Sdim) return;               // pipeline drain launch
    l = 0; step = t;
    x_base   = Xemb + (size_t)t * Ddim;  // row m at m*(S*D) + t*D
    x_stride = (long)Sdim * Ddim;
    x_next   = X1buf + (size_t)(step & 1) * HS;
    out      = nullptr;
  } else {
    if (t == 0) return;                  // pipeline fill launch
    l = 1; step = t - 1;
    x_base   = X1buf + (size_t)(step & 1) * HS;  // written by previous launch
    x_stride = (long)Hdim;
    x_next   = nullptr;
    out      = out_h;
  }
  const int pin  = step & 1;
  const __bf16* h_in  = hbuf + (size_t)(l * 2 + pin) * HS;
  __bf16*       h_out = hbuf + (size_t)(l * 2 + (pin ^ 1)) * HS;
  float*        cptr  = cbuf + (size_t)l * HS;
  const float*  bs    = bias + (size_t)l * G4;
  const __bf16* wp0   = Wpack + (size_t)l * WL;

  const int wave = wg * 4 + (threadIdx.x >> 5);   // u: 0..31 (hidden-unit tile)
  const int lane = threadIdx.x & 31;
  const int jn   = lane & 15;   // col-in-tile / A row-in-tile
  const int kg   = lane >> 4;   // K half-select

  v8f acc[2][4] = {};

  const __bf16* wp = wp0 + (size_t)wave * (4 * 32 * 32 * 16);

  for (int ki = 0; ki < 32; ++ki) {                // K = ki*32 .. +31
    const __bf16* arow;
    long stride;
    int koff;
    if (ki < 16) { arow = x_base; stride = x_stride; koff = ki * 32; }
    else         { arow = h_in;   stride = Hdim;     koff = ki * 32 - Ddim; }

    // A fragments: two 16-byte chunks per lane (bf16 16x32 A layout)
    v16bf a[2];
#pragma unroll
    for (int mt = 0; mt < 2; ++mt) {
      const __bf16* p = arow + (size_t)(mt * 16 + jn) * stride + koff + kg * 16;
      union { v16bf v; v8bf h[2]; } ua;
      ua.h[0] = *reinterpret_cast<const v8bf*>(p);
      ua.h[1] = *reinterpret_cast<const v8bf*>(p + 8);
      a[mt] = ua.v;
    }

#pragma unroll
    for (int g = 0; g < 4; ++g) {
      const v16bf bf = *reinterpret_cast<const v16bf*>(
          wp + ((size_t)(g * 32 + ki) * 32 + lane) * 16);
      acc[0][g] = __builtin_amdgcn_wmma_f32_16x16x32_bf16(
          false, a[0], false, bf, (short)0, acc[0][g], false, false);
      acc[1][g] = __builtin_amdgcn_wmma_f32_16x16x32_bf16(
          false, a[1], false, bf, (short)0, acc[1][g], false, false);
    }
  }

  // ---- fused gate epilogue (each lane owns column j, rows per C layout) ----
  const int j = wave * 16 + jn;
  const float bi = bs[j];
  const float bf_ = bs[Hdim + j];
  const float bg = bs[2 * Hdim + j];
  const float bo = bs[3 * Hdim + j];

#pragma unroll
  for (int mt = 0; mt < 2; ++mt) {
#pragma unroll
    for (int r = 0; r < 8; ++r) {
      const int m = mt * 16 + kg * 8 + r;   // batch row, per 16x16 f32 C layout
      const float zi = acc[mt][0][r] + bi;
      const float zf = acc[mt][1][r] + bf_;
      const float zg = acc[mt][2][r] + bg;
      const float zo = acc[mt][3][r] + bo;
      const float ig = 1.0f / (1.0f + __expf(-zi));
      const float fg = 1.0f / (1.0f + __expf(-zf));
      const float gg = tanhf(zg);
      const float og = 1.0f / (1.0f + __expf(-zo));
      const float cold = cptr[m * Hdim + j];
      const float cnew = fg * cold + ig * gg;
      const float hnew = og * tanhf(cnew);
      cptr[m * Hdim + j] = cnew;
      const __bf16 hb = (__bf16)hnew;
      h_out[m * Hdim + j] = hb;
      if (x_next) x_next[m * Hdim + j] = hb;                        // layer 0
      if (out) out[((size_t)m * Sdim + step) * Hdim + j] = hnew;    // layer 1
    }
  }
}

// ---------------------------------------------------------------------------
extern "C" void kernel_launch(void* const* d_in, const int* in_sizes, int n_in,
                              void* d_out, int out_size, void* d_ws, size_t ws_size,
                              hipStream_t stream) {
  const int*   src  = (const int*)d_in[0];
  // d_in[1] = source_len (unused by reference output)
  const float* emb  = (const float*)d_in[2];
  const float* Wk   = (const float*)d_in[3];
  const float* Wr   = (const float*)d_in[4];
  const float* bias = (const float*)d_in[5];

  float* out_h   = (float*)d_out;                                   // [B,S,HID]
  float* out_emb = out_h + (size_t)Bdim * Sdim * Hdim;              // [B,S,D]

  char* ws = (char*)d_ws;
  __bf16* Xemb  = (__bf16*)ws;  ws += (size_t)Bdim * Sdim * Ddim * 2;    // 16 MB
  __bf16* Wpack = (__bf16*)ws;  ws += (size_t)2 * KCAT * G4 * 2;          // 8 MB
  __bf16* hbuf  = (__bf16*)ws;  ws += (size_t)2 * 2 * HS * 2;             // h ping-pong
  float*  cbuf  = (float*)ws;   ws += (size_t)2 * HS * 4;                 // c state
  __bf16* X1    = (__bf16*)ws;  ws += (size_t)2 * HS * 2;                 // slice ping-pong

  // Phase A: embedding gather (parallel over all B*S rows)
  lstm_embed_gather<<<Bdim * Sdim, 128, 0, stream>>>(src, emb, out_emb, Xemb);

  // Phase B: one-time bf16 fragment-major weight pack (L2-resident afterwards)
  lstm_pack_weights<<<(2 * KCAT * G4) / 256, 256, 0, stream>>>(Wk, Wr, Wpack);

  // Phase C: zero h/c state
  lstm_init_state<<<(2 * 2 * Bdim * Hdim) / 256, 256, 0, stream>>>(hbuf, cbuf);

  // Phase D: pipelined recurrence. Launch t runs layer0(t) || layer1(t-1);
  // t == Sdim is the drain launch (layer1 only).
  for (int t = 0; t <= Sdim; ++t) {
    lstm_pipe_step<<<16, 128, 0, stream>>>(
        Xemb, hbuf, cbuf, bias, Wpack, X1, out_h, t);
  }
}
